// LinearAttentionBlock_58162447122669
// MI455X (gfx1250) — matrix operand order repaired and checked by reading, not compiled, not run on hardware
//
#include <hip/hip_runtime.h>
#include <hip/hip_bf16.h>

#define C_CH 256
#define NPIX 16384
#define B_SZ 8

typedef __attribute__((ext_vector_type(16))) __bf16 v16bf;
typedef __attribute__((ext_vector_type(8)))  float  v8f;

union Frag {
    v16bf v;
    uint4 q[2];
    unsigned u[8];
    unsigned short h[16];
};

__device__ __forceinline__ unsigned short f2bf(float f) {
    unsigned u = __float_as_uint(f);
    return (unsigned short)((u + 0x7FFFu + ((u >> 16) & 1u)) >> 16);   // RNE
}
__device__ __forceinline__ float bf2f(unsigned short h) {
    return __uint_as_float(((unsigned)h) << 16);
}
__device__ __forceinline__ v8f wmma_bf16(v16bf a, v16bf b, v8f c) {
    // D = A(16x32 bf16) * B(32x16 bf16) + C(16x16 f32)
    return __builtin_amdgcn_wmma_f32_16x16x32_bf16(false, a, false, b, (short)0, c, false, false);
}

// ---------------- K0: convert weights to bf16 ----------------
__global__ __launch_bounds__(256) void cvt_w(
    const float* __restrict__ Wq, const float* __restrict__ Wk, const float* __restrict__ Wv,
    unsigned short* __restrict__ dq, unsigned short* __restrict__ dk, unsigned short* __restrict__ dv)
{
    int i = blockIdx.x * 256 + threadIdx.x;   // 0..65535
    dq[i] = f2bf(Wq[i]);
    dk[i] = f2bf(Wk[i]);
    dv[i] = f2bf(Wv[i]);
}

// ---------------- K1: GroupNorm statistics ----------------
// Each (b,g) group is a contiguous slab of 32*16384 floats in x.
__global__ __launch_bounds__(256) void gn_stats(const float* __restrict__ x, float* __restrict__ stats)
{
    __shared__ float s1[256], s2[256];
    const float4* p = reinterpret_cast<const float4*>(x + (size_t)blockIdx.x * 32 * NPIX);
    float a = 0.f, b2 = 0.f;
    for (int i = threadIdx.x; i < 131072; i += 256) {
        float4 v = p[i];
        a  += v.x + v.y + v.z + v.w;
        b2 += v.x * v.x + v.y * v.y + v.z * v.z + v.w * v.w;
    }
    s1[threadIdx.x] = a; s2[threadIdx.x] = b2;
    __syncthreads();
    for (int off = 128; off > 0; off >>= 1) {
        if (threadIdx.x < off) {
            s1[threadIdx.x] += s1[threadIdx.x + off];
            s2[threadIdx.x] += s2[threadIdx.x + off];
        }
        __syncthreads();
    }
    if (threadIdx.x == 0) {
        const float inv = 1.0f / 524288.0f;
        float mean = s1[0] * inv;
        float var  = s2[0] * inv - mean * mean;
        stats[blockIdx.x]      = mean;
        stats[64 + blockIdx.x] = rsqrtf(var + 1e-5f);
    }
}

// ---------------- K2 helper: one 256x256 x (256 x 32px) WMMA GEMM ----------------
// mode 0: q -> elu+1 -> bf16 global (channel-contiguous). mode 1: k -> elu+1 -> LDS [ch][px].
// mode 2: v -> LDS [ch][px].
template<int MODE>
__device__ __forceinline__ void qkv_gemm(
    const unsigned short* __restrict__ xn_s,   // LDS [32px][264ch] bf16
    const unsigned short* __restrict__ Wb,     // bf16 256x256 row-major [out][in]
    const float* __restrict__ bias,
    unsigned short* lds_cm,                    // LDS [256ch][34px] (modes 1/2)
    unsigned short* __restrict__ qdst,         // q + ((b*NPIX+px0)*256) (mode 0)
    int tid)
{
    const int lane = tid & 31, wv = tid >> 5;
    const int lhi = lane >> 4, lm = lane & 15;
    v8f acc[2][2] = {};
#pragma unroll
    for (int ks = 0; ks < 8; ++ks) {
        const int k0a = ks * 32 + lhi * 16;    // A: lanes 0-15 K 0..15, lanes 16-31 K 16..31
        Frag a0, a1, b0, b1;
        a0.q[0] = *reinterpret_cast<const uint4*>(Wb + (wv * 32 + lm) * 256 + k0a);
        a0.q[1] = *reinterpret_cast<const uint4*>(Wb + (wv * 32 + lm) * 256 + k0a + 8);
        a1.q[0] = *reinterpret_cast<const uint4*>(Wb + (wv * 32 + 16 + lm) * 256 + k0a);
        a1.q[1] = *reinterpret_cast<const uint4*>(Wb + (wv * 32 + 16 + lm) * 256 + k0a + 8);
        const int k0b = ks * 32 + lhi * 16;    // B: lanes 0-15 K 0..15, lanes 16-31 K 16..31
        const unsigned* pb0 = reinterpret_cast<const unsigned*>(xn_s + lm * 264 + k0b);
        const unsigned* pb1 = reinterpret_cast<const unsigned*>(xn_s + (16 + lm) * 264 + k0b);
#pragma unroll
        for (int j = 0; j < 8; ++j) { b0.u[j] = pb0[j]; b1.u[j] = pb1[j]; }
        acc[0][0] = wmma_bf16(a0.v, b0.v, acc[0][0]);
        acc[0][1] = wmma_bf16(a0.v, b1.v, acc[0][1]);
        acc[1][0] = wmma_bf16(a1.v, b0.v, acc[1][0]);
        acc[1][1] = wmma_bf16(a1.v, b1.v, acc[1][1]);
    }
#pragma unroll
    for (int mt = 0; mt < 2; ++mt) {
#pragma unroll
        for (int nt = 0; nt < 2; ++nt) {
            const int cbase = wv * 32 + mt * 16 + lhi * 8;  // C/D: row = vgpr + 8*(lane>=16)
            const int pxl = nt * 16 + lm;                   // col = lane & 15
            if (MODE == 0) {
#pragma unroll
                for (int e = 0; e < 4; ++e) {
                    float v0 = acc[mt][nt][2 * e]     + bias[cbase + 2 * e];
                    float v1 = acc[mt][nt][2 * e + 1] + bias[cbase + 2 * e + 1];
                    v0 = v0 > 0.f ? v0 + 1.f : __expf(v0);  // elu(x)+1
                    v1 = v1 > 0.f ? v1 + 1.f : __expf(v1);
                    unsigned pk = (unsigned)f2bf(v0) | ((unsigned)f2bf(v1) << 16);
                    *reinterpret_cast<unsigned*>(qdst + (size_t)pxl * 256 + cbase + 2 * e) = pk;
                }
            } else {
#pragma unroll
                for (int j = 0; j < 8; ++j) {
                    float v0 = acc[mt][nt][j] + bias[cbase + j];
                    if (MODE == 1) v0 = v0 > 0.f ? v0 + 1.f : __expf(v0);
                    lds_cm[(cbase + j) * 34 + pxl] = f2bf(v0);
                }
            }
        }
    }
}

// ---------------- K2: fused GN-apply + QKV convs + kv/ksum accumulation ----------------
__global__ __launch_bounds__(256) void fused_qkv_kv(
    const float* __restrict__ x, const float* __restrict__ gamma, const float* __restrict__ beta,
    const unsigned short* __restrict__ Wqb, const float* __restrict__ bq,
    const unsigned short* __restrict__ Wkb, const float* __restrict__ bk,
    const unsigned short* __restrict__ Wvb, const float* __restrict__ bv,
    const float* __restrict__ stats,
    unsigned short* __restrict__ qws, float* __restrict__ kv, float* __restrict__ ksum)
{
    __shared__ unsigned short xn_s[32 * 264];   // [px][ch], padded stride (bank-conflict free)
    __shared__ unsigned short k_s[256 * 34];    // [ch][px]
    __shared__ unsigned short v_s[256 * 34];    // [ch][px]
    const int b = blockIdx.y;
    const int px0 = blockIdx.x * 32;
    const int tid = threadIdx.x;

    // Phase 1: load + normalize 256ch x 32px tile -> bf16 LDS
    {
        const int px4 = (tid & 7) * 4;
        const int csub = tid >> 3;
#pragma unroll
        for (int p = 0; p < 8; ++p) {
            const int c = p * 32 + csub;
            const int g = c >> 5;
            const float mean = stats[b * 8 + g];
            const float rvar = stats[64 + b * 8 + g];
            const float sc = rvar * gamma[c];
            const float sh = beta[c] - mean * sc;
            float4 xv = *reinterpret_cast<const float4*>(
                x + ((size_t)(b * C_CH + c)) * NPIX + px0 + px4);
            xn_s[(px4 + 0) * 264 + c] = f2bf(xv.x * sc + sh);
            xn_s[(px4 + 1) * 264 + c] = f2bf(xv.y * sc + sh);
            xn_s[(px4 + 2) * 264 + c] = f2bf(xv.z * sc + sh);
            xn_s[(px4 + 3) * 264 + c] = f2bf(xv.w * sc + sh);
        }
    }
    __syncthreads();

    // Phase 2: three WMMA GEMMs (each wave owns 32 output channels x 32 pixels)
    unsigned short* qdst = qws + ((size_t)b * NPIX + px0) * 256;
    qkv_gemm<0>(xn_s, Wqb, bq, nullptr, qdst, tid);
    qkv_gemm<1>(xn_s, Wkb, bk, k_s, nullptr, tid);
    qkv_gemm<2>(xn_s, Wvb, bv, v_s, nullptr, tid);
    __syncthreads();

    // Phase 3a: ksum (thread t owns channel t)
    {
        float s = 0.f;
#pragma unroll
        for (int p2 = 0; p2 < 32; ++p2) s += bf2f(k_s[tid * 34 + p2]);
        atomicAdd(&ksum[b * C_CH + tid], s);
    }

    // Phase 3b: kv partial = k(256 x 32px) * v^T(32px x 256), WMMA, atomic-accumulate (L2-resident)
    {
        const int lane = tid & 31, wv = tid >> 5;
        const int lhi = lane >> 4, lm = lane & 15;
        Frag a0, a1;
        const int k0a = lhi * 16;
        {
            const unsigned* p0 = reinterpret_cast<const unsigned*>(k_s + (wv * 32 + lm) * 34 + k0a);
            const unsigned* p1 = reinterpret_cast<const unsigned*>(k_s + (wv * 32 + 16 + lm) * 34 + k0a);
#pragma unroll
            for (int j = 0; j < 4; ++j) { a0.u[j] = p0[j]; a1.u[j] = p1[j]; }
            const unsigned* q0 = reinterpret_cast<const unsigned*>(k_s + (wv * 32 + lm) * 34 + k0a + 8);
            const unsigned* q1 = reinterpret_cast<const unsigned*>(k_s + (wv * 32 + 16 + lm) * 34 + k0a + 8);
#pragma unroll
            for (int j = 0; j < 4; ++j) { a0.u[4 + j] = q0[j]; a1.u[4 + j] = q1[j]; }
        }
        float* kvb = kv + (size_t)b * C_CH * C_CH;
        for (int nt2 = 0; nt2 < 16; ++nt2) {
            Frag bb;
            const unsigned* pb = reinterpret_cast<const unsigned*>(v_s + (nt2 * 16 + lm) * 34 + lhi * 16);
#pragma unroll
            for (int j = 0; j < 8; ++j) bb.u[j] = pb[j];
            v8f z = {};
            v8f r0 = wmma_bf16(a0.v, bb.v, z);
            v8f r1 = wmma_bf16(a1.v, bb.v, z);
            const int d = nt2 * 16 + lm;
#pragma unroll
            for (int j = 0; j < 8; ++j) {
                atomicAdd(&kvb[(size_t)(wv * 32 + lhi * 8 + j) * C_CH + d], r0[j]);
                atomicAdd(&kvb[(size_t)(wv * 32 + 16 + lhi * 8 + j) * C_CH + d], r1[j]);
            }
        }
    }
}

// ---------------- K3: M_b = Wp @ (kv / (ksum + eps)) -> bf16 ----------------
__global__ __launch_bounds__(256) void proj_mat(
    const float* __restrict__ Wp, const float* __restrict__ kv, const float* __restrict__ ksum,
    unsigned short* __restrict__ Mb)
{
    __shared__ float wp_s[16 * 256];
    __shared__ float rinv[256];
    const int b = blockIdx.y, ot = blockIdx.x, tid = threadIdx.x;
#pragma unroll
    for (int r = 0; r < 16; ++r) wp_s[r * 256 + tid] = Wp[(ot * 16 + r) * 256 + tid];
    rinv[tid] = 1.0f / (ksum[b * 256 + tid] + 1e-6f);
    __syncthreads();
    float acc[16];
#pragma unroll
    for (int o = 0; o < 16; ++o) acc[o] = 0.f;
    const float* kvb = kv + (size_t)b * 65536;
    for (int c = 0; c < 256; ++c) {
        float kd = kvb[c * 256 + tid] * rinv[c];
#pragma unroll
        for (int o = 0; o < 16; ++o) acc[o] += wp_s[o * 256 + c] * kd;
    }
#pragma unroll
    for (int o = 0; o < 16; ++o)
        Mb[(size_t)(b * 256 + ot * 16 + o) * 256 + tid] = f2bf(acc[o]);
}

// ---------------- K4: out = M_b @ q + bp + x (q tile staged once in LDS) ----------------
__global__ __launch_bounds__(256) void out_gemm(
    const unsigned short* __restrict__ Mb, const unsigned short* __restrict__ qws,
    const float* __restrict__ x, const float* __restrict__ bp, float* __restrict__ out)
{
    __shared__ unsigned short q_s[64 * 264];   // [px][ch] bf16, padded stride
    const int b = blockIdx.y, px0 = blockIdx.x * 64, tid = threadIdx.x;
    const int lane = tid & 31, wv = tid >> 5, lhi = lane >> 4, lm = lane & 15;
    const unsigned short* M = Mb + (size_t)b * 65536;
    const unsigned short* q = qws + ((size_t)b * NPIX + px0) * 256;

    // Stage 64px x 256ch q tile -> LDS (coalesced uint4 copies, ds_store_b128)
#pragma unroll
    for (int it = 0; it < 8; ++it) {
        const int flat = it * 256 + tid;      // 0..2047 uint4 chunks
        const int px = flat >> 5;             // 32 uint4 per pixel row
        const int c8 = (flat & 31) * 8;       // 8 bf16 per uint4
        *reinterpret_cast<uint4*>(&q_s[px * 264 + c8]) =
            *reinterpret_cast<const uint4*>(q + (size_t)px * 256 + c8);
    }
    __syncthreads();

    v8f acc[2][4] = {};
#pragma unroll
    for (int ks = 0; ks < 8; ++ks) {
        const int k0a = ks * 32 + lhi * 8;
        Frag a0, a1;
        a0.q[0] = *reinterpret_cast<const uint4*>(M + (wv * 32 + lm) * 256 + k0a);
        a0.q[1] = *reinterpret_cast<const uint4*>(M + (wv * 32 + lm) * 256 + k0a + 16);
        a1.q[0] = *reinterpret_cast<const uint4*>(M + (wv * 32 + 16 + lm) * 256 + k0a);
        a1.q[1] = *reinterpret_cast<const uint4*>(M + (wv * 32 + 16 + lm) * 256 + k0a + 16);
        const int k0b = ks * 32 + lhi * 8;
#pragma unroll
        for (int nt = 0; nt < 4; ++nt) {
            Frag bb;
            const unsigned short* ql = &q_s[(nt * 16 + lm) * 264 + k0b];
            bb.q[0] = *reinterpret_cast<const uint4*>(ql);
            bb.q[1] = *reinterpret_cast<const uint4*>(ql + 16);
            acc[0][nt] = wmma_bf16(a0.v, bb.v, acc[0][nt]);
            acc[1][nt] = wmma_bf16(a1.v, bb.v, acc[1][nt]);
        }
    }
#pragma unroll
    for (int mt = 0; mt < 2; ++mt) {
#pragma unroll
        for (int nt = 0; nt < 4; ++nt) {
            const int cbase = wv * 32 + mt * 16 + lhi * 8;
            const int pxg = px0 + nt * 16 + lm;
#pragma unroll
            for (int j = 0; j < 8; ++j) {
                const size_t idx = (size_t)(b * C_CH + cbase + j) * NPIX + pxg;
                out[idx] = acc[mt][nt][j] + bp[cbase + j] + x[idx];
            }
        }
    }
}

// ---------------- host launcher ----------------
extern "C" void kernel_launch(void* const* d_in, const int* in_sizes, int n_in,
                              void* d_out, int out_size, void* d_ws, size_t ws_size,
                              hipStream_t stream) {
    const float* x     = (const float*)d_in[0];
    const float* gamma = (const float*)d_in[1];
    const float* beta  = (const float*)d_in[2];
    const float* Wq    = (const float*)d_in[3];
    const float* bq    = (const float*)d_in[4];
    const float* Wk    = (const float*)d_in[5];
    const float* bk    = (const float*)d_in[6];
    const float* Wv    = (const float*)d_in[7];
    const float* bv    = (const float*)d_in[8];
    const float* Wp    = (const float*)d_in[9];
    const float* bp    = (const float*)d_in[10];
    float* out = (float*)d_out;
    char* ws = (char*)d_ws;

    const size_t OFF_Q     = 0;                                           // q bf16: b*n*c
    const size_t OFF_WQ    = OFF_Q + (size_t)B_SZ * NPIX * C_CH * 2;
    const size_t OFF_WK    = OFF_WQ + (size_t)65536 * 2;
    const size_t OFF_WV    = OFF_WK + (size_t)65536 * 2;
    const size_t OFF_M     = OFF_WV + (size_t)65536 * 2;                  // M_b bf16: b*c*c
    const size_t OFF_KV    = OFF_M + (size_t)B_SZ * 65536 * 2;            // kv f32
    const size_t OFF_KSUM  = OFF_KV + (size_t)B_SZ * 65536 * 4;           // ksum f32
    const size_t OFF_STATS = OFF_KSUM + (size_t)B_SZ * C_CH * 4;          // mean[64], rvar[64]

    unsigned short* wsq   = (unsigned short*)(ws + OFF_WQ);
    unsigned short* wsk   = (unsigned short*)(ws + OFF_WK);
    unsigned short* wsv   = (unsigned short*)(ws + OFF_WV);
    unsigned short* qws   = (unsigned short*)(ws + OFF_Q);
    unsigned short* Mb    = (unsigned short*)(ws + OFF_M);
    float* kvp   = (float*)(ws + OFF_KV);
    float* ksump = (float*)(ws + OFF_KSUM);
    float* stats = (float*)(ws + OFF_STATS);

    cvt_w<<<256, 256, 0, stream>>>(Wq, Wk, Wv, wsq, wsk, wsv);
    gn_stats<<<64, 256, 0, stream>>>(x, stats);
    hipMemsetAsync(ws + OFF_KV, 0, (size_t)B_SZ * 65536 * 4 + (size_t)B_SZ * C_CH * 4, stream);
    fused_qkv_kv<<<dim3(NPIX / 32, B_SZ), 256, 0, stream>>>(
        x, gamma, beta, wsq, bq, wsk, bk, wsv, bv, stats, qws, kvp, ksump);
    proj_mat<<<dim3(16, B_SZ), 256, 0, stream>>>(Wp, kvp, ksump, Mb);
    out_gemm<<<dim3(NPIX / 64, B_SZ), 256, 0, stream>>>(Mb, qws, x, bp, out);
}
